// TinyAttention_8340826489600
// MI455X (gfx1250) — matrix operand order repaired and checked by reading, not compiled, not run on hardware
//
#include <hip/hip_runtime.h>

// ---------------------------------------------------------------------------
// TinyAttention for MI455X (gfx1250, wave32, WMMA bf16 -> f32 accumulate)
//   K0a: pack W_qkv f32 -> bf16 [kblock][col][kk]   (B-fragment-linear)
//   K0b: pack W_proj f32 -> bf16 transposed [2048][64]
//   K1 : qkv = x @ W_qkv + b_qkv   (TDM-staged W panel in LDS, bf16 out;
//                                   v-part stored transposed vT[b][d][n])
//   K2 : flash attention per (batch, 16-row q tile), ctx bf16 (ws)
//   K3 : out = ctx @ W_proj + b_proj  (f32 out)
// ---------------------------------------------------------------------------

typedef __bf16 bf16;
typedef __attribute__((ext_vector_type(16))) __bf16        v16bf;
typedef __attribute__((ext_vector_type(8)))  float         v8f;
typedef __attribute__((ext_vector_type(4)))  unsigned int  u32x4;
typedef __attribute__((ext_vector_type(8)))  int           i32x8;
typedef __attribute__((ext_vector_type(4)))  int           i32x4;

#define BATCH  8
#define NSEQ   2048
#define DIN    2048
#define DA     64
#define QKVC   192          // 3 * DA
#define DOUT   2048
#define WPANEL (32 * QKVC)  // 6144 elements = 12 KB per k-step panel

static __device__ __forceinline__ v8f wmma_bf16(v16bf a, v16bf b, v8f c) {
  // emits v_wmma_f32_16x16x32_bf16
  return __builtin_amdgcn_wmma_f32_16x16x32_bf16(false, a, false, b,
                                                 (short)0, c, false, false);
}

// ---- WMMA wave32 fragment loaders (CDNA5 ISA 7.12.2 register layouts) -----

// A (16M x 32K) fragment; src -> element (0,0) of tile, row-major, ld elems.
static __device__ __forceinline__ v16bf load_a_bf16(const bf16* src, int ld) {
  const int lane = threadIdx.x & 31;
  const int m = lane & 15, h = lane >> 4;
  const bf16* p = src + (size_t)m * ld + h * 16;
  v16bf a;
#pragma unroll
  for (int j = 0; j < 8; ++j) {
    const int k = j << 1;  // 0,2,4,...,14
    a[2 * j]     = p[k];
    a[2 * j + 1] = p[k + 1];
  }
  return a;
}

// B = Tᵀ for row-major T (row n of T becomes column n of B); src -> T(0, k0).
// Per-lane contiguous 32B; all call sites are 32B-aligned -> vector load.
static __device__ __forceinline__ v16bf load_bt_bf16(const bf16* src, int ld) {
  const int lane = threadIdx.x & 31;
  const int n = lane & 15, h = lane >> 4;
  return *(const v16bf*)(src + (size_t)n * ld + h * 16);
}

// B fragment from packed-linear panel: element (k,n) at panel[(c0+n)*32 + k].
static __device__ __forceinline__ v16bf load_b_panel(const bf16* panel, int c0) {
  const int lane = threadIdx.x & 31;
  const int n = lane & 15, h = lane >> 4;
  return *(const v16bf*)(panel + (size_t)(c0 + n) * 32 + h * 16);
}

// B fragment from Wᵀ[2048][64]: column n of W = row (c0+n) of Wᵀ.
static __device__ __forceinline__ v16bf load_b_wpT(const bf16* WpT, int c0,
                                                   int kchunk) {
  const int lane = threadIdx.x & 31;
  const int n = lane & 15, h = lane >> 4;
  return *(const v16bf*)(WpT + (size_t)(c0 + n) * DA + kchunk * 32 + h * 16);
}

// ---- TDM: async linear panel load (global bf16 -> LDS), ISA ch.8 D# -------
// packed W tensor seen as 2D [64 rows][6144 cols]; tile = 1 row x 6144.
static __device__ __forceinline__ void tdm_load_wpanel(unsigned lds_addr,
                                                       const bf16* gaddr) {
  const unsigned long long ga = (unsigned long long)gaddr;
  u32x4 g0;
  g0[0] = 1u;                                   // count=1, user descriptor
  g0[1] = lds_addr;                             // lds_addr (bytes)
  g0[2] = (unsigned)ga;                         // global_addr[31:0]
  g0[3] = (unsigned)((ga >> 32) & 0x01FFFFFFu)  // global_addr[56:32]
          | 0x80000000u;                        // type=2 ("image")
  i32x8 g1;
  g1[0] = 1 << 16;                              // data_size=1 (2 bytes)
  g1[1] = WPANEL << 16;                         // tensor_dim0 = 6144 (lo16)
  g1[2] = 64 << 16;                             // tensor_dim1 = 64 (lo16)
  g1[3] = WPANEL << 16;                         // tile_dim0 = 6144
  g1[4] = 1;                                    // tile_dim1 = 1; tile_dim2=0
  g1[5] = WPANEL;                               // tensor_dim0_stride lo32
  g1[6] = 0;
  g1[7] = 0;
  const i32x4 z4 = {0, 0, 0, 0};                // groups 2/3 unused (2D)
  const i32x8 z8 = {0, 0, 0, 0, 0, 0, 0, 0};
  __builtin_amdgcn_tensor_load_to_lds(g0, g1, z4, z4, z8, 0);
}

// ---------------------------------------------------------------------------
// K0a: W_qkv[2048][192] f32 -> packed bf16: dst[(k/32)*192 + c][k%32]
// ---------------------------------------------------------------------------
__global__ __launch_bounds__(256)
void pack_wqkv_kernel(const float* __restrict__ src, bf16* __restrict__ dst) {
  const int i = blockIdx.x * 256 + threadIdx.x;
  if (i >= DIN * QKVC) return;
  const int k = i / QKVC, c = i % QKVC;
  dst[((size_t)(k >> 5) * QKVC + c) * 32 + (k & 31)] = (bf16)src[i];
}

// ---------------------------------------------------------------------------
// K0b: W_proj[64][2048] f32 -> bf16 transposed dst[c][k]
// ---------------------------------------------------------------------------
__global__ __launch_bounds__(256)
void pack_wp_kernel(const float* __restrict__ src, bf16* __restrict__ dst) {
  const int i = blockIdx.x * 256 + threadIdx.x;
  if (i >= DA * DOUT) return;
  const int k = i / DOUT, c = i % DOUT;
  dst[(size_t)c * DA + k] = (bf16)src[i];
}

// ---------------------------------------------------------------------------
// K1: qkv = x @ W_qkv + b_qkv. 4 waves/WG; WG owns 16 rows x all 192 cols.
// x tile (16x32) converted once into LDS, shared by all 12 column tiles;
// packed W panel (12KB) streamed into LDS by the Tensor Data Mover,
// double-buffered, synchronized with s_wait_tensorcnt + barriers.
// q,k parts -> qkv row-major; v part -> vT[b][d][n] for contiguous PV reads.
// ---------------------------------------------------------------------------
__global__ __launch_bounds__(128)
void qkv_gemm_kernel(const float* __restrict__ x, const bf16* __restrict__ Wpk,
                     const float* __restrict__ bqkv, bf16* __restrict__ qkv,
                     bf16* __restrict__ vT) {
  __shared__ __align__(32) bf16 xtile[16 * 32];       // 1 KB
  __shared__ __align__(32) bf16 wpanel[2][WPANEL];    // 2 x 12 KB (TDM dest)
  const int rt   = blockIdx.x;                  // 0..1023
  const int tid  = threadIdx.x;
  const int wave = tid >> 5;                    // 0..3
  const int lane = tid & 31;
  const int n = lane & 15, h = lane >> 4;
  const int r0 = rt * 16;

  v8f acc[3];                                   // col tiles c0 = wave*48+c*16
#pragma unroll
  for (int c = 0; c < 3; ++c) {
    const float bias = bqkv[wave * 48 + c * 16 + n];
#pragma unroll
    for (int r = 0; r < 8; ++r) acc[c][r] = bias;
  }

  // prologue: TDM panel 0
  if (wave == 0) {
    tdm_load_wpanel((unsigned)(unsigned long long)(void*)&wpanel[0][0], Wpk);
    __builtin_amdgcn_s_wait_tensorcnt(0);
  }
  __syncthreads();

  int cur = 0;
  for (int k0 = 0; k0 < DIN; k0 += 32) {
    // ---- stage x tile: 128 threads x float4, convert to bf16 in LDS ----
    {
      const int e = tid * 4;                    // 512 elements total
      const int xm = e >> 5, xc = e & 31;
      const float4 xv =
          *(const float4*)(x + (size_t)(r0 + xm) * DIN + k0 + xc);
      bf16* dst = &xtile[xm * 32 + xc];
      dst[0] = (bf16)xv.x; dst[1] = (bf16)xv.y;
      dst[2] = (bf16)xv.z; dst[3] = (bf16)xv.w;
    }
    // ---- kick TDM for next W panel into the other buffer ----
    const bool have_next = (k0 + 32) < DIN;
    if (wave == 0 && have_next)
      tdm_load_wpanel((unsigned)(unsigned long long)(void*)&wpanel[cur ^ 1][0],
                      Wpk + (size_t)((k0 >> 5) + 1) * WPANEL);
    __syncthreads();                            // xtile ready

    // ---- 12 WMMAs per workgroup (3 per wave), shared A fragment ----
    const v16bf a = load_a_bf16(xtile, 32);
#pragma unroll
    for (int c = 0; c < 3; ++c) {
      const v16bf b = load_b_panel(&wpanel[cur][0], wave * 48 + c * 16);
      acc[c] = wmma_bf16(a, b, acc[c]);
    }
    __syncthreads();                            // done with xtile/wpanel[cur]
    if (wave == 0 && have_next)
      __builtin_amdgcn_s_wait_tensorcnt(0);     // next panel landed
    __syncthreads();
    cur ^= 1;
  }

  const int bb    = r0 / NSEQ;                  // batch of this row block
  const int rowin = r0 - bb * NSEQ;
#pragma unroll
  for (int c = 0; c < 3; ++c) {
    const int c0 = wave * 48 + c * 16;
#pragma unroll
    for (int r = 0; r < 8; ++r) {
      const int m = r + h * 8;
      const bf16 val = (bf16)acc[c][r];
      if (c0 < 2 * DA)                          // q,k: row-major
        qkv[(size_t)(r0 + m) * QKVC + c0 + n] = val;
      else                                      // v: transposed vT[b][d][row]
        vT[((size_t)bb * DA + (c0 - 2 * DA) + n) * NSEQ + rowin + m] = val;
    }
  }
}

// ---------------------------------------------------------------------------
// K2: flash attention. One wave = one (batch, 16-row q tile). Streams keys in
// blocks of 32; scores live in C-layout accumulators (lane = key column), so
// per-q-row softmax stats reduce with shfl_xor {1,2,4,8} inside 16-lane halves.
// P goes C-layout -> LDS -> A-layout, then 4 WMMAs accumulate the 16x64 ctx.
// K fragments read contiguous from qkv, V fragments contiguous from vT.
// ---------------------------------------------------------------------------
__global__ __launch_bounds__(32)
void attn_kernel(const bf16* __restrict__ qkv, const bf16* __restrict__ vT,
                 bf16* __restrict__ ctxbuf) {
  __shared__ __align__(32) bf16 pbuf[16 * 32];   // P tile bounce (1 KB)
  const int qt = blockIdx.x;                     // 0..127
  const int b  = blockIdx.y;                     // 0..7
  const int lane = threadIdx.x & 31;
  const int nn = lane & 15, h = lane >> 4;
  const size_t rowbase = (size_t)b * NSEQ;
  const int r0 = qt * 16;

  const bf16* qbase = qkv + (rowbase + r0) * QKVC;          // q at col 0
  const v16bf aq0 = load_a_bf16(qbase, QKVC);
  const v16bf aq1 = load_a_bf16(qbase + 32, QKVC);
  const bf16* vtb = vT + (size_t)b * DA * NSEQ;

  v8f octx[4];
#pragma unroll
  for (int t = 0; t < 4; ++t) octx[t] = (v8f){};
  float mrow[8], lrow[8];
#pragma unroll
  for (int r = 0; r < 8; ++r) { mrow[r] = -1e30f; lrow[r] = 0.f; }

  const float scale = 0.125f;                    // rsqrt(64)

  for (int j0 = 0; j0 < NSEQ; j0 += 32) {
    const bf16* kb0 = qkv + (rowbase + j0) * QKVC + DA;       // k at col 64
    const bf16* kb1 = qkv + (rowbase + j0 + 16) * QKVC + DA;
    if (j0 + 32 < NSEQ)                          // global_prefetch_b8
      __builtin_prefetch(kb0 + (size_t)32 * QKVC + (lane & 15) * QKVC, 0, 3);

    v8f s0 = (v8f){}, s1 = (v8f){};
    s0 = wmma_bf16(aq0, load_bt_bf16(kb0, QKVC), s0);
    s0 = wmma_bf16(aq1, load_bt_bf16(kb0 + 32, QKVC), s0);
    s1 = wmma_bf16(aq0, load_bt_bf16(kb1, QKVC), s1);
    s1 = wmma_bf16(aq1, load_bt_bf16(kb1 + 32, QKVC), s1);

    float alpha[8];
#pragma unroll
    for (int r = 0; r < 8; ++r) {
      s0[r] *= scale; s1[r] *= scale;
      float v = fmaxf(s0[r], s1[r]);
      v = fmaxf(v, __shfl_xor(v, 1, 32));
      v = fmaxf(v, __shfl_xor(v, 2, 32));
      v = fmaxf(v, __shfl_xor(v, 4, 32));
      v = fmaxf(v, __shfl_xor(v, 8, 32));
      const float mn = fmaxf(mrow[r], v);
      alpha[r] = __expf(mrow[r] - mn);
      mrow[r] = mn;
      s0[r] = __expf(s0[r] - mn);
      s1[r] = __expf(s1[r] - mn);
      float sum = s0[r] + s1[r];
      sum += __shfl_xor(sum, 1, 32);
      sum += __shfl_xor(sum, 2, 32);
      sum += __shfl_xor(sum, 4, 32);
      sum += __shfl_xor(sum, 8, 32);
      lrow[r] = lrow[r] * alpha[r] + sum;
    }

    // P: C-layout -> LDS (16 x 32 row-major) -> A-layout
#pragma unroll
    for (int r = 0; r < 8; ++r) {
      const int m = r + h * 8;
      pbuf[m * 32 + nn]      = (bf16)s0[r];
      pbuf[m * 32 + 16 + nn] = (bf16)s1[r];
    }
    __syncthreads();
    const v16bf ap = load_a_bf16(pbuf, 32);
    __syncthreads();

#pragma unroll
    for (int t = 0; t < 4; ++t)
#pragma unroll
      for (int r = 0; r < 8; ++r) octx[t][r] *= alpha[r];

    // V fragments: lane n holds vT row (t*16+n), keys j0..j0+31 contiguous
#pragma unroll
    for (int t = 0; t < 4; ++t)
      octx[t] = wmma_bf16(ap, load_bt_bf16(vtb + (size_t)(t * 16) * NSEQ + j0,
                                           NSEQ), octx[t]);
  }

#pragma unroll
  for (int r = 0; r < 8; ++r) lrow[r] = 1.0f / lrow[r];
  bf16* crow = ctxbuf + (rowbase + r0) * DA;
#pragma unroll
  for (int t = 0; t < 4; ++t)
#pragma unroll
    for (int r = 0; r < 8; ++r) {
      const int m = r + h * 8;
      crow[(size_t)m * DA + t * 16 + nn] = (bf16)(octx[t][r] * lrow[r]);
    }
}

// ---------------------------------------------------------------------------
// K3: out[16384][2048] (f32) = ctx[16384][64] @ W_proj + b_proj  (Wᵀ packed)
// ---------------------------------------------------------------------------
__global__ __launch_bounds__(32)
void out_proj_kernel(const bf16* __restrict__ ctxbuf, const bf16* __restrict__ WpT,
                     const float* __restrict__ bp, float* __restrict__ out) {
  const int ct = blockIdx.x;            // 0..127
  const int rt = blockIdx.y;            // 0..1023
  const int lane = threadIdx.x & 31;
  const int n = lane & 15, h = lane >> 4;
  const int r0 = rt * 16, c0 = ct * 16;

  v8f acc;
  const float bias = bp[c0 + n];
#pragma unroll
  for (int r = 0; r < 8; ++r) acc[r] = bias;

  const v16bf a0 = load_a_bf16(ctxbuf + (size_t)r0 * DA, DA);
  const v16bf a1 = load_a_bf16(ctxbuf + (size_t)r0 * DA + 32, DA);
  acc = wmma_bf16(a0, load_b_wpT(WpT, c0, 0), acc);
  acc = wmma_bf16(a1, load_b_wpT(WpT, c0, 1), acc);

#pragma unroll
  for (int r = 0; r < 8; ++r)
    out[(size_t)(r0 + r + h * 8) * DOUT + c0 + n] = acc[r];
}

// ---------------------------------------------------------------------------
extern "C" void kernel_launch(void* const* d_in, const int* in_sizes, int n_in,
                              void* d_out, int out_size, void* d_ws, size_t ws_size,
                              hipStream_t stream) {
  const float* x    = (const float*)d_in[0];   // [8,2048,2048]
  const float* Wqkv = (const float*)d_in[1];   // [2048,192]
  const float* bqkv = (const float*)d_in[2];   // [192]
  const float* Wp   = (const float*)d_in[3];   // [64,2048]
  const float* bp   = (const float*)d_in[4];   // [2048]
  float* out = (float*)d_out;                  // [8,2048,2048]

  // ws (bf16): qkv 6.29MB | vT 2.10MB | ctx 2.10MB | Wpk 0.79MB | WpT 0.26MB
  bf16* qkv  = (bf16*)d_ws;
  bf16* vT   = qkv  + (size_t)BATCH * NSEQ * QKVC;
  bf16* ctxb = vT   + (size_t)BATCH * NSEQ * DA;
  bf16* Wpk  = ctxb + (size_t)BATCH * NSEQ * DA;
  bf16* WpT  = Wpk  + (size_t)DIN * QKVC;

  pack_wqkv_kernel<<<(DIN * QKVC + 255) / 256, 256, 0, stream>>>(Wqkv, Wpk);
  pack_wp_kernel<<<(DA * DOUT + 255) / 256, 256, 0, stream>>>(Wp, WpT);

  qkv_gemm_kernel<<<dim3((BATCH * NSEQ) / 16), 128, 0, stream>>>(
      x, Wpk, bqkv, qkv, vT);
  attn_kernel<<<dim3(NSEQ / 16, BATCH), 32, 0, stream>>>(qkv, vT, ctxb);
  out_proj_kernel<<<dim3(DOUT / 16, (BATCH * NSEQ) / 16), 32, 0, stream>>>(
      ctxb, WpT, bp, out);
}
